// RNN_71167608094732
// MI455X (gfx1250) — matrix operand order repaired and checked by reading, not compiled, not run on hardware
//
#include <hip/hip_runtime.h>
#include <hip/hip_bf16.h>

// ---------------------------------------------------------------------------
// RNN decode on MI455X (gfx1250): bf16 WMMA (v_wmma_f32_16x16x32_bf16) +
// Tensor Data Mover (tensor_load_to_lds / tensor_store_from_lds) for the
// recurrent h-state tiles, tracked with TENSORcnt.
// ---------------------------------------------------------------------------

constexpr int L_  = 4;
constexpr int H_  = 1024;
constexpr int B_  = 64;
constexpr int T_  = 512;
constexpr int IN_ = 128;
constexpr int OUT_= 128;

typedef __attribute__((ext_vector_type(16))) __bf16 v16bf;
typedef __attribute__((ext_vector_type(8)))  float  v8f;
typedef __attribute__((ext_vector_type(4)))  unsigned int v4u;
typedef __attribute__((ext_vector_type(8)))  int    v8i;
typedef __attribute__((ext_vector_type(4)))  int    v4i;

__device__ __forceinline__ v8f wmma_bf16(v16bf a, v16bf b, v8f c) {
  return __builtin_amdgcn_wmma_f32_16x16x32_bf16(false, a, false, b,
                                                 (short)0, c, false, false);
}

// A fragment from row-major bf16 (LDS): p points at row base + k0.
__device__ __forceinline__ v16bf ld_a(const __bf16* p) {
  union { uint4 u[2]; v16bf v; } r;
  r.u[0] = *(const uint4*)(p);        // K e0..7
  r.u[1] = *(const uint4*)(p + 16);   // K e8..15 (k0+16..k0+23)
  return r.v;
}

// B fragment from prepacked weights: tile (kt, nt), lane-contiguous 32 B.
__device__ __forceinline__ v16bf ld_b(const __bf16* wb, int kt, int nt,
                                      int NT, int lane) {
  const __bf16* p = wb + ((((size_t)kt * NT) + nt) * 32 + lane) * 16;
  union { uint4 u[2]; v16bf v; } r;
  r.u[0] = *(const uint4*)(p);
  r.u[1] = *(const uint4*)(p + 8);
  return r.v;
}

// ---------------------------------------------------------------------------
// Tensor DMA descriptor for a 2-D bf16 tile: 16 rows x 1024 elements, row
// stride 1024 elements, global <-> contiguous LDS.  (CDNA5 ISA §8.3/8.4)
// ---------------------------------------------------------------------------
__device__ __forceinline__ void tdm_desc(v4u& g0, v8i& g1,
                                         unsigned lds_addr,
                                         unsigned long long gaddr) {
  g0[0] = 1u;                                          // count=1, user mode
  g0[1] = lds_addr;                                    // LDS byte address
  g0[2] = (unsigned)(gaddr & 0xffffffffu);             // global_addr[31:0]
  g0[3] = (unsigned)((gaddr >> 32) & 0x01ffffffu)      // global_addr[56:32]
        | 0x80000000u;                                 // type = 2 ("image")
  g1[0] = 0x00010000;        // workgroup_mask=0, data_size=1 (2 bytes)
  g1[1] = (int)(1024u << 16);// tensor_dim0[15:0]=1024 in bits[63:48]
  g1[2] = (int)(16u << 16);  // tensor_dim0[31:16]=0, tensor_dim1[15:0]=16
  g1[3] = (int)(1024u << 16);// tensor_dim1[31:16]=0, tile_dim0=1024
  g1[4] = 16;                // tile_dim1=16, tile_dim2=0
  g1[5] = 1024;              // tensor_dim0_stride[31:0] = 1024 elements
  g1[6] = 0;                 // stride hi / tensor_dim1_stride lo
  g1[7] = 0;
}

__device__ __forceinline__ void tdm_load_tile(unsigned lds_addr,
                                              unsigned long long gaddr) {
  v4u g0; v8i g1; tdm_desc(g0, g1, lds_addr, gaddr);
  v4i z4 = {0, 0, 0, 0};
  v8i z8 = {0, 0, 0, 0, 0, 0, 0, 0};
  __builtin_amdgcn_tensor_load_to_lds(g0, g1, z4, z4, z8, 0);
}

__device__ __forceinline__ void tdm_store_tile(unsigned lds_addr,
                                               unsigned long long gaddr) {
  v4u g0; v8i g1; tdm_desc(g0, g1, lds_addr, gaddr);
  v4i z4 = {0, 0, 0, 0};
  v8i z8 = {0, 0, 0, 0, 0, 0, 0, 0};
  __builtin_amdgcn_tensor_store_from_lds(g0, g1, z4, z4, z8, 0);
}

// ---------------------------------------------------------------------------
// Prologue: convert fp32 weight [K][N] (row-major) into WMMA B-fragment
// tile layout: dst[((kt*NT+nt)*32 + lane)*16 + e] = bf16(W[k][n]).
// ---------------------------------------------------------------------------
__global__ void prepack_w(const float* __restrict__ W, __bf16* __restrict__ dst,
                          int K, int N) {
  const int NT = N >> 4;
  const long total = (long)K * N;
  for (long i = (long)blockIdx.x * blockDim.x + threadIdx.x; i < total;
       i += (long)gridDim.x * blockDim.x) {
    const int  e    = (int)(i & 15);
    const int  lane = (int)((i >> 4) & 31);
    const long tile = i >> 9;
    const int  nt   = (int)(tile % NT);
    const int  kt   = (int)(tile / NT);
    const int  n = nt * 16 + (lane & 15);
    const int  k = kt * 32 + ((e >> 3) << 4) + ((lane >> 4) << 3) + (e & 7);
    dst[i] = (__bf16)W[(long)k * N + n];
  }
}

// Fuse the duplicated biases; zero both h-state ping-pong buffers.
__global__ void prologue_misc(const float* __restrict__ bhd, const float* __restrict__ bh,
                              const float* __restrict__ byd, const float* __restrict__ by,
                              const float* __restrict__ bod, const float* __restrict__ ob,
                              float* __restrict__ bhc, float* __restrict__ byc,
                              float* __restrict__ boc, __bf16* __restrict__ hst) {
  const int i = blockIdx.x * blockDim.x + threadIdx.x;
  if (i < L_ * H_) {
    bhc[i] = bhd[i] + bh[i];
    byc[i] = byd[i] + by[i];
  }
  if (i < OUT_) boc[i] = bod[i] + ob[i];
  const int tot = 2 * L_ * B_ * H_;
  for (int j = i; j < tot; j += gridDim.x * blockDim.x) hst[j] = (__bf16)0.0f;
}

// ---------------------------------------------------------------------------
// One timestep. Grid = 4 workgroups (each owns 16 batch rows), 512 threads
// (16 waves). Each wave owns 4 N-tiles per GEMM with a shared A fragment.
//   xin[16][1024] = layer input / residual carrier (LDS resident all step)
//   xh [16][1024] = h_prev tile (TDM load), then h tile (TDM store)
// ---------------------------------------------------------------------------
__global__ __launch_bounds__(512, 1) void rnn_step(
    int t,
    const float*  __restrict__ x,     // [B][T][IN] fp32
    const __bf16* __restrict__ WiB,   // KT=4,  NT=64
    const float*  __restrict__ bi,    // [H]
    const __bf16* __restrict__ WhB,   // [L] KT=64, NT=64
    const float*  __restrict__ bhc,   // [L][H]
    const __bf16* __restrict__ WyB,   // [L] KT=32, NT=64
    const float*  __restrict__ byc,   // [L][H]
    const __bf16* __restrict__ WoB,   // KT=32, NT=8
    const float*  __restrict__ boc,   // [OUT]
    __bf16*       __restrict__ hst,   // [2][L][B][H]
    float*        __restrict__ out)   // [B][T][OUT]
{
  __shared__ __bf16 xin[16][1024];    // 32 KB
  __shared__ __bf16 xh [16][1024];    // 32 KB

  const int tid  = threadIdx.x;
  const int lane = tid & 31;
  const int w    = tid >> 5;          // wave 0..15
  const int hi   = lane >> 4;         // 0/1 lane half
  const int ml   = lane & 15;
  const int rowbase = blockIdx.x * 16;

  const int p = t & 1;
  const __bf16* hr = hst + (size_t)p       * (L_ * B_ * H_);
  __bf16*       hw = hst + (size_t)(p ^ 1) * (L_ * B_ * H_);
  const unsigned xh_lds = (unsigned)(size_t)(void*)&xh[0][0];

  // ---- GEMM0: inp = x[:,t,:] @ Wi + bi   (K=128, fp32->bf16 on the fly)
  {
    v8f acc[4];
    #pragma unroll
    for (int i = 0; i < 4; ++i) {
      const float bv = bi[(w * 4 + i) * 16 + ml];
      #pragma unroll
      for (int v = 0; v < 8; ++v) acc[i][v] = bv;
    }
    const float* xrow = x + ((size_t)(rowbase + ml) * T_ + t) * IN_;
    #pragma unroll
    for (int kt = 0; kt < 4; ++kt) {
      const int k0 = kt * 32 + hi * 8;
      const float4 f0 = *(const float4*)(xrow + k0);
      const float4 f1 = *(const float4*)(xrow + k0 + 4);
      const float4 f2 = *(const float4*)(xrow + k0 + 16);
      const float4 f3 = *(const float4*)(xrow + k0 + 20);
      v16bf a;
      a[0]=(__bf16)f0.x;  a[1]=(__bf16)f0.y;  a[2]=(__bf16)f0.z;  a[3]=(__bf16)f0.w;
      a[4]=(__bf16)f1.x;  a[5]=(__bf16)f1.y;  a[6]=(__bf16)f1.z;  a[7]=(__bf16)f1.w;
      a[8]=(__bf16)f2.x;  a[9]=(__bf16)f2.y;  a[10]=(__bf16)f2.z; a[11]=(__bf16)f2.w;
      a[12]=(__bf16)f3.x; a[13]=(__bf16)f3.y; a[14]=(__bf16)f3.z; a[15]=(__bf16)f3.w;
      #pragma unroll
      for (int i = 0; i < 4; ++i)
        acc[i] = wmma_bf16(a, ld_b(WiB, kt, w * 4 + i, 64, lane), acc[i]);
    }
    #pragma unroll
    for (int i = 0; i < 4; ++i) {
      const int nb = (w * 4 + i) * 16 + ml;
      #pragma unroll
      for (int v = 0; v < 8; ++v) xin[v + 8 * hi][nb] = (__bf16)acc[i][v];
    }
  }

  // ---- layer loop
  for (int m = 0; m < L_; ++m) {
    __syncthreads();
    // TDM: h_prev[m] rows rowbase..rowbase+15 -> xh (32 KB, async DMA).
    if (w == 0) {
      const unsigned long long ga =
          (unsigned long long)(size_t)(hr + ((size_t)m * B_ + rowbase) * H_);
      tdm_load_tile(xh_lds, ga);
      __builtin_amdgcn_s_wait_tensorcnt(0);   // also drains prior store
    }
    __syncthreads();

    // GEMM1: h = relu(concat(inp, h_prev) @ Wh[m] + bhc[m]); K = 2048.
    v8f acc[4];
    {
      const __bf16* Wm = WhB + (size_t)m * (64 * 64 * 512);
      #pragma unroll
      for (int i = 0; i < 4; ++i) {
        const float bv = bhc[m * H_ + (w * 4 + i) * 16 + ml];
        #pragma unroll
        for (int v = 0; v < 8; ++v) acc[i][v] = bv;
      }
      for (int kt = 0; kt < 32; ++kt) {             // K 0..1023 from xin
        const v16bf a = ld_a(&xin[ml][kt * 32 + hi * 8]);
        #pragma unroll
        for (int i = 0; i < 4; ++i)
          acc[i] = wmma_bf16(a, ld_b(Wm, kt, w * 4 + i, 64, lane), acc[i]);
      }
      for (int kt = 0; kt < 32; ++kt) {             // K 1024..2047 from xh
        const v16bf a = ld_a(&xh[ml][kt * 32 + hi * 8]);
        #pragma unroll
        for (int i = 0; i < 4; ++i)
          acc[i] = wmma_bf16(a, ld_b(Wm, kt + 32, w * 4 + i, 64, lane), acc[i]);
      }
    }
    __syncthreads();                  // all A-reads of xh done
    #pragma unroll
    for (int i = 0; i < 4; ++i) {     // overwrite h_prev tile with h
      const int nb = (w * 4 + i) * 16 + ml;
      #pragma unroll
      for (int v = 0; v < 8; ++v)
        xh[v + 8 * hi][nb] = (__bf16)fmaxf(acc[i][v], 0.0f);
    }
    __syncthreads();                  // xh == h

    // TDM: persist h tile for next timestep (async LDS -> global).
    if (w == 0) {
      const unsigned long long ga =
          (unsigned long long)(size_t)(hw + ((size_t)m * B_ + rowbase) * H_);
      tdm_store_tile(xh_lds, ga);     // drained by next layer's wait / endpgm
    }

    // GEMM2: out = h @ Wy[m] + byc[m] + 2*inp; K = 1024. Writes back inp.
    {
      const __bf16* Wm = WyB + (size_t)m * (32 * 64 * 512);
      v8f a2[4];
      #pragma unroll
      for (int i = 0; i < 4; ++i) {
        const float bv = byc[m * H_ + (w * 4 + i) * 16 + ml];
        #pragma unroll
        for (int v = 0; v < 8; ++v) a2[i][v] = bv;
      }
      for (int kt = 0; kt < 32; ++kt) {
        const v16bf a = ld_a(&xh[ml][kt * 32 + hi * 8]);
        #pragma unroll
        for (int i = 0; i < 4; ++i)
          a2[i] = wmma_bf16(a, ld_b(Wm, kt, w * 4 + i, 64, lane), a2[i]);
      }
      #pragma unroll
      for (int i = 0; i < 4; ++i) {
        const int nb = (w * 4 + i) * 16 + ml;
        #pragma unroll
        for (int v = 0; v < 8; ++v) {
          const float o = a2[i][v] + 2.0f * (float)xin[v + 8 * hi][nb];
          xin[v + 8 * hi][nb] = (__bf16)o;   // becomes next layer's inp
        }
      }
    }
  }

  __syncthreads();

  // ---- o_proj: out[:,t,:] = dec @ Wo + boc  (K=1024, N=128 -> 8 tiles)
  if (w < 8) {
    v8f acc;
    const float bv = boc[w * 16 + ml];
    #pragma unroll
    for (int v = 0; v < 8; ++v) acc[v] = bv;
    for (int kt = 0; kt < 32; ++kt) {
      const v16bf a = ld_a(&xin[ml][kt * 32 + hi * 8]);
      acc = wmma_bf16(a, ld_b(WoB, kt, w, 8, lane), acc);
    }
    #pragma unroll
    for (int v = 0; v < 8; ++v)
      out[((size_t)(rowbase + v + 8 * hi) * T_ + t) * OUT_ + w * 16 + ml] = acc[v];
  }
  // s_endpgm performs an implicit wait-idle, draining the last tensor store.
}

// ---------------------------------------------------------------------------
extern "C" void kernel_launch(void* const* d_in, const int* in_sizes, int n_in,
                              void* d_out, int out_size, void* d_ws, size_t ws_size,
                              hipStream_t stream) {
  (void)in_sizes; (void)n_in; (void)out_size; (void)ws_size;
  const float* x   = (const float*)d_in[0];
  const float* Wi  = (const float*)d_in[1];
  const float* bi  = (const float*)d_in[2];
  const float* Wh  = (const float*)d_in[3];
  const float* bhd = (const float*)d_in[4];
  const float* bh  = (const float*)d_in[5];
  const float* Wy  = (const float*)d_in[6];
  const float* byd = (const float*)d_in[7];
  const float* by  = (const float*)d_in[8];
  const float* Wo  = (const float*)d_in[9];
  const float* bod = (const float*)d_in[10];
  const float* ob  = (const float*)d_in[11];
  float* out = (float*)d_out;

  // Workspace layout (≈ 25.5 MB total, all offsets 512 B-multiples).
  char* ws = (char*)d_ws;
  __bf16* WiB = (__bf16*)ws; ws += 262144;     //  4*64 tiles * 1 KB
  __bf16* WhB = (__bf16*)ws; ws += 16777216;   //  4 layers * 64*64 tiles
  __bf16* WyB = (__bf16*)ws; ws += 8388608;    //  4 layers * 32*64 tiles
  __bf16* WoB = (__bf16*)ws; ws += 262144;     //  32*8 tiles
  __bf16* hst = (__bf16*)ws; ws += 1048576;    //  2 * L * B * H bf16
  float*  bhc = (float*)ws;  ws += 16384;
  float*  byc = (float*)ws;  ws += 16384;
  float*  boc = (float*)ws;  ws += 512;

  prepack_w<<<256, 256, 0, stream>>>(Wi, WiB, 128, 1024);
  for (int m = 0; m < L_; ++m)
    prepack_w<<<1024, 256, 0, stream>>>(Wh + (size_t)m * 2048 * 1024,
                                        WhB + (size_t)m * 2097152, 2048, 1024);
  for (int m = 0; m < L_; ++m)
    prepack_w<<<512, 256, 0, stream>>>(Wy + (size_t)m * 1024 * 1024,
                                       WyB + (size_t)m * 1048576, 1024, 1024);
  prepack_w<<<256, 256, 0, stream>>>(Wo, WoB, 1024, 128);
  prologue_misc<<<1024, 256, 0, stream>>>(bhd, bh, byd, by, bod, ob,
                                          bhc, byc, boc, hst);

  for (int t = 0; t < T_; ++t)
    rnn_step<<<4, 512, 0, stream>>>(t, x, WiB, bi, WhB, bhc, WyB, byc,
                                    WoB, boc, hst, out);
}
